// EncoderLayer_33990371180965
// MI455X (gfx1250) — compile-verified, matrix operand order checked by reading, not run on hardware
//
#include <hip/hip_runtime.h>
#include <cmath>

// ---------- types ----------
typedef _Float16 f16;
typedef __attribute__((ext_vector_type(16))) _Float16 v16h;
typedef __attribute__((ext_vector_type(8)))  _Float16 v8h;
typedef __attribute__((ext_vector_type(8)))  float    v8f;

__device__ __forceinline__ v8f wmma_f16(v16h a, v16h b, v8f c) {
    // D(16x16,f32) = A(16x32,f16) x B(32x16,f16) + C
    return __builtin_amdgcn_wmma_f32_16x16x32_f16(
        /*neg_a=*/false, a, /*neg_b=*/false, b,
        /*c_mod=*/(short)0, c, /*reuse_a=*/false, /*reuse_b=*/false);
}

__device__ __forceinline__ v16h frag_from(const f16* lo_base, const f16* hi_base) {
    v8h lo = *(const v8h*)lo_base;
    v8h hi = *(const v8h*)hi_base;
    v16h r;
    #pragma unroll
    for (int e = 0; e < 8; ++e) { r[e] = lo[e]; r[8 + e] = hi[e]; }
    return r;
}

// ---------- fp32 -> fp16 conversion ----------
__global__ void cvt_f32_to_f16_kernel(const float* __restrict__ src,
                                      f16* __restrict__ dst, int n) {
    int i = blockIdx.x * blockDim.x + threadIdx.x;
    if (i < n) dst[i] = (f16)src[i];
}

// ---------- LayerNorm (torch style: alpha*(x-mu)/(std+eps)+bias, ddof=1) ----------
__global__ void layernorm_f16_kernel(const float* __restrict__ x,
                                     const float* __restrict__ alpha,
                                     const float* __restrict__ beta,
                                     f16* __restrict__ out, int D) {
    const int row = blockIdx.x;
    const float* xr = x + (size_t)row * D;
    float s = 0.f, s2 = 0.f;
    for (int c = threadIdx.x; c < D; c += blockDim.x) {
        float v = xr[c];
        s += v; s2 += v * v;
    }
    __shared__ float rs[256], rq[256];
    rs[threadIdx.x] = s; rq[threadIdx.x] = s2;
    __syncthreads();
    for (int off = 128; off > 0; off >>= 1) {
        if ((int)threadIdx.x < off) {
            rs[threadIdx.x] += rs[threadIdx.x + off];
            rq[threadIdx.x] += rq[threadIdx.x + off];
        }
        __syncthreads();
    }
    const float mu  = rs[0] / (float)D;
    const float var = (rq[0] - (float)D * mu * mu) / (float)(D - 1);
    const float inv = 1.0f / (sqrtf(var) + 1e-6f);
    f16* orow = out + (size_t)row * D;
    for (int c = threadIdx.x; c < D; c += blockDim.x)
        orow[c] = (f16)(alpha[c] * (xr[c] - mu) * inv + beta[c]);
}

// ---------- generic WMMA GEMM: C[M,N] = A[M,K] * W[N,K]^T + bias ----------
// Block tile 128x128, BK=32; 8 waves, each owns a 32x64 patch (2x4 WMMAs/step).
// Template-selected epilogue: ReLU / f32 residual+out / f16 out / f16 transposed out.
#define BM 128
#define BN 128
#define BK 32
#define LDSP 40   // padded LDS row stride in halfs (80B, 16B-aligned rows)

template <bool RELU, bool RES, bool OUTF, bool OUTH, bool OUTHT>
__global__ void __launch_bounds__(256)
gemm_wmma_kernel(const f16* __restrict__ A, const f16* __restrict__ W,
                 const float* __restrict__ bias, const float* __restrict__ residual,
                 float* __restrict__ outF, f16* __restrict__ outH,
                 f16* __restrict__ outHT, int M, int N, int K, int Srows) {
    __shared__ __align__(16) f16 sA[BM * LDSP];
    __shared__ __align__(16) f16 sB[BN * LDSP];

    const int t      = threadIdx.x;
    const int lane   = t & 31;
    const int wave   = t >> 5;           // 0..7
    const int wm     = wave & 3;         // 4 row groups of 32 rows
    const int wn     = wave >> 2;        // 2 col groups of 64 cols
    const int half16 = lane >> 4;
    const int l16    = lane & 15;

    const int bm = blockIdx.y * BM;
    const int bn = blockIdx.x * BN;

    // loaders: 128 rows x (2 x 16 halfs) for each of A, B
    const int lrow = t >> 1, lhalf = t & 1;
    const f16* gA = A + (size_t)(bm + lrow) * K + lhalf * 16;
    const f16* gW = W + (size_t)(bn + lrow) * K + lhalf * 16;
    f16* sArow = &sA[lrow * LDSP + lhalf * 16];
    f16* sBrow = &sB[lrow * LDSP + lhalf * 16];

    v8f acc[2][4] = {};

    for (int kk = 0; kk < K; kk += BK) {
        v8h a0 = *(const v8h*)(gA + kk);
        v8h a1 = *(const v8h*)(gA + kk + 8);
        v8h b0 = *(const v8h*)(gW + kk);
        v8h b1 = *(const v8h*)(gW + kk + 8);
        if (kk + BK < K) {  // CDNA5 global_prefetch of next K tile
            __builtin_prefetch(gA + kk + BK, 0, 0);
            __builtin_prefetch(gW + kk + BK, 0, 0);
        }
        *(v8h*)(sArow)     = a0;
        *(v8h*)(sArow + 8) = a1;
        *(v8h*)(sBrow)     = b0;
        *(v8h*)(sBrow + 8) = b1;
        __syncthreads();

        v16h af[2], bf[4];
        #pragma unroll
        for (int i = 0; i < 2; ++i) {
            const f16* ap = &sA[(wm * 32 + i * 16 + l16) * LDSP];
            af[i] = frag_from(ap + half16 * 8, ap + 16 + half16 * 8);
        }
        #pragma unroll
        for (int j = 0; j < 4; ++j) {
            const f16* bp = &sB[(wn * 64 + j * 16 + l16) * LDSP];
            bf[j] = frag_from(bp + half16 * 8, bp + 16 + half16 * 8);
        }
        #pragma unroll
        for (int i = 0; i < 2; ++i)
            #pragma unroll
            for (int j = 0; j < 4; ++j)
                acc[i][j] = wmma_f16(af[i], bf[j], acc[i][j]);
        __syncthreads();
    }

    // epilogue: C layout -> lane holds col l16, rows half16*8 + r
    #pragma unroll
    for (int i = 0; i < 2; ++i)
        #pragma unroll
        for (int j = 0; j < 4; ++j) {
            const int r0 = bm + wm * 32 + i * 16 + half16 * 8;
            const int c  = bn + wn * 64 + j * 16 + l16;
            const float bv = bias[c];
            #pragma unroll
            for (int r = 0; r < 8; ++r) {
                const int m = r0 + r;
                float v = acc[i][j][r] + bv;
                if (RELU) v = fmaxf(v, 0.f);
                const size_t idx = (size_t)m * N + c;
                if (RES)  v += residual[idx];
                if (OUTF) outF[idx] = v;
                if (OUTH) outH[idx] = (f16)v;
                if (OUTHT) {
                    const int bb = m / Srows, ss = m % Srows;
                    outHT[((size_t)bb * N + c) * Srows + ss] = (f16)v;
                }
            }
        }
}

// ---------- flash attention ----------
// 4 waves / block; each wave owns one 16-query tile for one (b,h).
// Q,K row-major f16 [B*S, 1024]; V pre-transposed: VT[b, d, s] contiguous in s.
__global__ void __launch_bounds__(128)
attention_kernel(const f16* __restrict__ Q, const f16* __restrict__ K,
                 const f16* __restrict__ VT, const int* __restrict__ mask,
                 f16* __restrict__ O) {
    constexpr int S = 2048, D = 1024, DK = 64;
    const int lane   = threadIdx.x & 31;
    const int wid    = threadIdx.x >> 5;
    const int half16 = lane >> 4;
    const int l16    = lane & 15;
    const int q0 = (blockIdx.x * 4 + wid) * 16;
    const int h  = blockIdx.y;
    const int b  = blockIdx.z;

    // Q fragments (A layout): row = l16, K subsets {h8..h8+7, 16+h8..16+h8+7}
    v16h qa[2];
    const f16* qrow = Q + (size_t)(b * S + q0 + l16) * D + h * DK;
    #pragma unroll
    for (int f = 0; f < 2; ++f)
        qa[f] = frag_from(qrow + f * 32 + half16 * 8, qrow + f * 32 + 16 + half16 * 8);

    float m_i[8], l_i[8];
    v8f oacc[4] = {};
    #pragma unroll
    for (int r = 0; r < 8; ++r) { m_i[r] = -INFINITY; l_i[r] = 0.f; }

    __shared__ __align__(16) f16 ldsP[4][16 * 33];
    const float scale = 0.125f;  // 1/sqrt(64)

    for (int j = 0; j < S; j += 32) {
        // scores for 32 keys as two 16x16 tiles
        v8f s[2] = {};
        int mk[2];
        #pragma unroll
        for (int tt = 0; tt < 2; ++tt) {
            mk[tt] = mask[(size_t)b * S + j + tt * 16 + l16];
            const f16* krow = K + (size_t)(b * S + j + tt * 16 + l16) * D + h * DK;
            #pragma unroll
            for (int f = 0; f < 2; ++f) {
                v16h kb = frag_from(krow + f * 32 + half16 * 8,
                                    krow + f * 32 + 16 + half16 * 8);
                s[tt] = wmma_f16(qa[f], kb, s[tt]);
            }
        }
        // online softmax (row stats redundant within each 16-lane half)
        float mnew[8], corr[8];
        #pragma unroll
        for (int r = 0; r < 8; ++r) {
            float v0 = s[0][r] * scale; if (mk[0] == 1) v0 = -INFINITY;
            float v1 = s[1][r] * scale; if (mk[1] == 1) v1 = -INFINITY;
            s[0][r] = v0; s[1][r] = v1;
            float mx = fmaxf(v0, v1);
            mx = fmaxf(mx, __shfl_xor(mx, 1, 32));
            mx = fmaxf(mx, __shfl_xor(mx, 2, 32));
            mx = fmaxf(mx, __shfl_xor(mx, 4, 32));
            mx = fmaxf(mx, __shfl_xor(mx, 8, 32));
            mnew[r] = fmaxf(m_i[r], mx);
            corr[r] = (m_i[r] == -INFINITY) ? 0.f : __expf(m_i[r] - mnew[r]);
        }
        #pragma unroll
        for (int r = 0; r < 8; ++r) {
            float p0 = (s[0][r] == -INFINITY) ? 0.f : __expf(s[0][r] - mnew[r]);
            float p1 = (s[1][r] == -INFINITY) ? 0.f : __expf(s[1][r] - mnew[r]);
            s[0][r] = p0; s[1][r] = p1;
            float sm = p0 + p1;
            sm += __shfl_xor(sm, 1, 32);
            sm += __shfl_xor(sm, 2, 32);
            sm += __shfl_xor(sm, 4, 32);
            sm += __shfl_xor(sm, 8, 32);
            l_i[r] = l_i[r] * corr[r] + sm;
            m_i[r] = mnew[r];
        }
        #pragma unroll
        for (int n = 0; n < 4; ++n)
            #pragma unroll
            for (int r = 0; r < 8; ++r) oacc[n][r] *= corr[r];

        // P: C layout -> LDS -> A-fragment layout (per-wave slab)
        __syncthreads();
        #pragma unroll
        for (int tt = 0; tt < 2; ++tt)
            #pragma unroll
            for (int r = 0; r < 8; ++r)
                ldsP[wid][(half16 * 8 + r) * 33 + tt * 16 + l16] = (f16)s[tt][r];
        __syncthreads();
        v16h pa;
        #pragma unroll
        for (int i = 0; i < 8; ++i) {
            pa[i]     = ldsP[wid][l16 * 33 + half16 * 8 + i];
            pa[8 + i] = ldsP[wid][l16 * 33 + 16 + half16 * 8 + i];
        }
        // O(16x64) += P(16x32) x V(32x64); VT gives contiguous key runs
        #pragma unroll
        for (int n = 0; n < 4; ++n) {
            const f16* vt = VT + ((size_t)b * D + h * DK + n * 16 + l16) * S + j;
            v16h vb = frag_from(vt + half16 * 8, vt + 16 + half16 * 8);
            oacc[n] = wmma_f16(pa, vb, oacc[n]);
        }
        __syncthreads();
    }

    #pragma unroll
    for (int n = 0; n < 4; ++n)
        #pragma unroll
        for (int r = 0; r < 8; ++r) {
            float val = oacc[n][r] / l_i[r];
            O[(size_t)(b * S + q0 + half16 * 8 + r) * D + h * DK + n * 16 + l16] = (f16)val;
        }
}

// ---------- host orchestration ----------
extern "C" void kernel_launch(void* const* d_in, const int* in_sizes, int n_in,
                              void* d_out, int out_size, void* d_ws, size_t ws_size,
                              hipStream_t stream) {
    (void)in_sizes; (void)n_in; (void)out_size; (void)ws_size;
    constexpr int Bn = 4, S = 2048, D = 1024, H = 16, DFF = 2048;
    constexpr int M = Bn * S;  // 8192 rows

    const float* x      = (const float*)d_in[0];
    const int*   mask   = (const int*)  d_in[1];
    const float* alpha1 = (const float*)d_in[2];
    const float* bias1  = (const float*)d_in[3];
    const float* alpha2 = (const float*)d_in[4];
    const float* bias2  = (const float*)d_in[5];
    const float* Wq = (const float*)d_in[6];  const float* bq = (const float*)d_in[7];
    const float* Wk = (const float*)d_in[8];  const float* bk = (const float*)d_in[9];
    const float* Wv = (const float*)d_in[10]; const float* bv = (const float*)d_in[11];
    const float* Wo = (const float*)d_in[12]; const float* bo = (const float*)d_in[13];
    const float* W1 = (const float*)d_in[14]; const float* b1 = (const float*)d_in[15];
    const float* W2 = (const float*)d_in[16]; const float* b2 = (const float*)d_in[17];

    char* ws = (char*)d_ws;
    size_t off = 0;
    auto alloc_h = [&](size_t elems) -> f16* {
        f16* p = (f16*)(ws + off); off += elems * sizeof(f16); return p;
    };
    f16* wq16 = alloc_h((size_t)D * D);
    f16* wk16 = alloc_h((size_t)D * D);
    f16* wv16 = alloc_h((size_t)D * D);
    f16* wo16 = alloc_h((size_t)D * D);
    f16* w116 = alloc_h((size_t)DFF * D);
    f16* w216 = alloc_h((size_t)D * DFF);
    f16* a16  = alloc_h((size_t)M * DFF);   // norm1 out (M*D) then FFN hidden (M*DFF)
    f16* q16  = alloc_h((size_t)M * D);     // Q, later norm2 out
    f16* k16  = alloc_h((size_t)M * D);
    f16* vt16 = alloc_h((size_t)M * D);     // V transposed: [b, d, s]
    f16* o16  = alloc_h((size_t)M * D);     // attention concat
    off = (off + 255) & ~(size_t)255;
    float* y32 = (float*)(ws + off);        // x + attention residual (fp32)

    const int CT = 256;
    // 1) weights fp32 -> fp16
    cvt_f32_to_f16_kernel<<<(D * D + CT - 1) / CT, CT, 0, stream>>>(Wq, wq16, D * D);
    cvt_f32_to_f16_kernel<<<(D * D + CT - 1) / CT, CT, 0, stream>>>(Wk, wk16, D * D);
    cvt_f32_to_f16_kernel<<<(D * D + CT - 1) / CT, CT, 0, stream>>>(Wv, wv16, D * D);
    cvt_f32_to_f16_kernel<<<(D * D + CT - 1) / CT, CT, 0, stream>>>(Wo, wo16, D * D);
    cvt_f32_to_f16_kernel<<<(DFF * D + CT - 1) / CT, CT, 0, stream>>>(W1, w116, DFF * D);
    cvt_f32_to_f16_kernel<<<(D * DFF + CT - 1) / CT, CT, 0, stream>>>(W2, w216, D * DFF);

    // 2) norm1: x -> a16 (f16)
    layernorm_f16_kernel<<<M, 256, 0, stream>>>(x, alpha1, bias1, a16, D);

    // 3) Q/K projections (f16 out); V projection writes transposed VT
    dim3 gqkv(D / BN, M / BM);
    gemm_wmma_kernel<false, false, false, true, false><<<gqkv, 256, 0, stream>>>(
        a16, wq16, bq, nullptr, nullptr, q16, nullptr, M, D, D, S);
    gemm_wmma_kernel<false, false, false, true, false><<<gqkv, 256, 0, stream>>>(
        a16, wk16, bk, nullptr, nullptr, k16, nullptr, M, D, D, S);
    gemm_wmma_kernel<false, false, false, false, true><<<gqkv, 256, 0, stream>>>(
        a16, wv16, bv, nullptr, nullptr, nullptr, vt16, M, D, D, S);

    // 4) flash attention -> o16 (f16 concat, heads are contiguous 64-col slices)
    attention_kernel<<<dim3(S / 16 / 4, H, Bn), 128, 0, stream>>>(q16, k16, vt16, mask, o16);

    // 5) output projection + residual: y32 = x + o16 @ Wo^T + bo
    gemm_wmma_kernel<false, true, true, false, false><<<gqkv, 256, 0, stream>>>(
        o16, wo16, bo, x, y32, nullptr, nullptr, M, D, D, S);

    // 6) norm2: y32 -> q16 (reused)
    layernorm_f16_kernel<<<M, 256, 0, stream>>>(y32, alpha2, bias2, q16, D);

    // 7) FFN1 (ReLU) -> a16 (M x DFF, f16)
    gemm_wmma_kernel<true, false, false, true, false><<<dim3(DFF / BN, M / BM), 256, 0, stream>>>(
        q16, w116, b1, nullptr, nullptr, a16, nullptr, M, DFF, D, S);

    // 8) FFN2 + residual -> d_out (fp32)
    gemm_wmma_kernel<false, true, true, false, false><<<dim3(D / BN, M / BM), 256, 0, stream>>>(
        a16, w216, b2, y32, (float*)d_out, nullptr, nullptr, M, D, DFF, S);
}
